// MultiHeadSelfAttention_35665408426651
// MI455X (gfx1250) — matrix operand order repaired and checked by reading, not compiled, not run on hardware
//
#include <hip/hip_runtime.h>
#include <hip/hip_bf16.h>

typedef __bf16 bf16_t;
typedef __attribute__((ext_vector_type(16))) __bf16 bf16x16;
typedef __attribute__((ext_vector_type(8)))  __bf16 bf16x8;
typedef __attribute__((ext_vector_type(8)))  float  f32x8;
typedef __attribute__((ext_vector_type(4)))  unsigned int u32x4;
typedef __attribute__((ext_vector_type(8)))  int i32x8;
typedef __attribute__((ext_vector_type(4)))  int i32x4;

#define B_   4
#define T_   2048
#define D_   1024
#define H_   16
#define DH_  64

static __device__ __forceinline__ f32x8 wmma_bf16(bf16x16 a, bf16x16 b, f32x8 c) {
  // D = A(16x32 bf16) * B(32x16 bf16) + C(16x16 f32)
  return __builtin_amdgcn_wmma_f32_16x16x32_bf16(false, a, false, b, (short)0, c, false, false);
}

// ---- Tensor Data Mover: 2D tile global->LDS (D# per CDNA5 ISA 8.3/8.4) ----
// 6-arg toolchain form: (u32x4 g0, i32x8 g1, i32x4, i32x4, i32x8, i32 cpol)
static __device__ __forceinline__ void tdm_load_2d(
    unsigned lds_byte_off, const void* gptr,
    unsigned td0, unsigned td1, unsigned long long stride0,
    unsigned tile0, unsigned tile1) {
  u32x4 g0;
  i32x8 g1;
  i32x4 gz4 = {0, 0, 0, 0};
  i32x8 gz8 = {0, 0, 0, 0, 0, 0, 0, 0};
  unsigned long long ga = (unsigned long long)(size_t)gptr;
  g0[0] = 1u;                                   // count=1, is_restore=0, gather off
  g0[1] = lds_byte_off;                         // lds_addr [63:32]
  g0[2] = (unsigned)ga;                         // global_addr[31:0]
  g0[3] = (unsigned)(ga >> 32) | (2u << 30);    // global_addr[56:32] | type=2
  g1[0] = 1 << 16;                              // workgroup_mask=0, data_size=1 (2B)
  g1[1] = (int)((td0 & 0xffffu) << 16);         // tensor_dim0[15:0]
  g1[2] = (int)((td0 >> 16) | ((td1 & 0xffffu) << 16));
  g1[3] = (int)((td1 >> 16) | (tile0 << 16));   // tensor_dim1[31:16] | tile_dim0
  g1[4] = (int)tile1;                           // tile_dim1, tile_dim2=0
  g1[5] = (int)(unsigned)stride0;               // tensor_dim0_stride[31:0]
  g1[6] = (int)(unsigned)((stride0 >> 32) & 0xffffu);
  g1[7] = 0;
  __builtin_amdgcn_tensor_load_to_lds(g0, g1, gz4, gz4, gz8, 0);
}

// ---------------- conversion kernels ----------------

__global__ __launch_bounds__(256) void cvt_kernel(const float* __restrict__ in,
                                                  bf16_t* __restrict__ out, int n) {
  int i = blockIdx.x * 256 + threadIdx.x;
  if (i < n) out[i] = (bf16_t)in[i];
}

// W[1024][colsN] fp32 -> WT[colsN][1024] bf16  (B-operand friendly layout)
__global__ __launch_bounds__(256) void cvtT_kernel(const float* __restrict__ in,
                                                   bf16_t* __restrict__ out, int colsN) {
  int i = blockIdx.x * 256 + threadIdx.x;
  if (i < colsN * D_) {
    int k = i & (D_ - 1);
    int n = i >> 10;
    out[i] = (bf16_t)in[(size_t)k * colsN + n];
  }
}

// ---------------- QKV projection GEMM (wave tile 16x64) ----------------
__global__ __launch_bounds__(256) void qkv_gemm_kernel(
    const bf16_t* __restrict__ xb, const bf16_t* __restrict__ wT,
    const float* __restrict__ bias,
    bf16_t* __restrict__ Q, bf16_t* __restrict__ K, bf16_t* __restrict__ Vt) {
  const int lane = threadIdx.x & 31, wv = threadIdx.x >> 5;
  const int half = lane >> 4, lm = lane & 15;
  const int n0 = (blockIdx.x * 8 + wv) * 64;   // 0..3071, step 64
  const int m0 = blockIdx.y * 16;              // 0..8191
  const bf16_t* arow = xb + (size_t)(m0 + lm) * D_;
  const bf16_t* brow = wT + (size_t)(n0 + lm) * D_;

  f32x8 acc[4] = {{}, {}, {}, {}};
  for (int kk = 0; kk < D_; kk += 32) {
    bf16x8 alo = *(const bf16x8*)(arow + kk + half * 16);
    bf16x8 ahi = *(const bf16x8*)(arow + kk + half * 16 + 8);
    bf16x16 a;
#pragma unroll
    for (int j = 0; j < 8; ++j) { a[j] = alo[j]; a[8 + j] = ahi[j]; }
    __builtin_prefetch(arow + kk + 128, 0, 3);
#pragma unroll
    for (int cb = 0; cb < 4; ++cb) {
      bf16x16 b = *(const bf16x16*)(brow + (size_t)cb * 16 * D_ + kk + half * 16);
      acc[cb] = wmma_bf16(a, b, acc[cb]);
    }
  }

#pragma unroll
  for (int cb = 0; cb < 4; ++cb) {
    const int col = n0 + cb * 16 + lm;
    const float bv = bias[col];
    const int sect = col >> 10;        // 0=Q 1=K 2=V
    const int cl = col & (D_ - 1);
    const int h = cl >> 6, d = cl & (DH_ - 1);
#pragma unroll
    for (int r = 0; r < 8; ++r) {
      int mrow = m0 + half * 8 + r;
      int bb = mrow >> 11, t = mrow & (T_ - 1);
      bf16_t v = (bf16_t)(acc[cb][r] + bv);
      size_t qk = ((size_t)(bb * H_ + h) * T_ + t) * DH_ + d;
      if (sect == 0)      Q[qk] = v;
      else if (sect == 1) K[qk] = v;
      else                Vt[((size_t)(bb * H_ + h) * DH_ + d) * T_ + t] = v;
    }
  }
}

// ---------------- flash attention with TDM staging ----------------
// 4 waves/block share one (b,h); wave w owns query tile qt0+w.
// K/V chunks (32 kv positions) double-buffered in LDS via tensor_load_to_lds.
__global__ __launch_bounds__(128) void attn_kernel(
    const bf16_t* __restrict__ Q, const bf16_t* __restrict__ K,
    const bf16_t* __restrict__ Vt, bf16_t* __restrict__ O) {
  // [ K buf0 | K buf1 | V buf0 | V buf1 | P scratch x4 waves ]
  __shared__ __attribute__((aligned(128))) bf16_t smem[4 * 2048 + 4 * 512];
  const int lane = threadIdx.x & 31, wv = threadIdx.x >> 5;
  const int half = lane >> 4, lm = lane & 15;
  const int wg = blockIdx.x * 4 + wv;
  const int qt = wg & (T_ / 16 - 1);
  const int bh = wg >> 7;

  bf16_t* sK0 = smem;                 // 2 x 32x64
  bf16_t* sV0 = smem + 4096;          // 2 x 64x32
  bf16_t* sP  = smem + 8192 + wv * 512;
  const unsigned lds_base = (unsigned)(size_t)(void*)smem;  // flat low 32 = LDS addr

  // Q fragments (A operand), rows qt*16+lm, dh chunks 0..31 / 32..63
  const bf16_t* qbase = Q + ((size_t)bh * T_ + qt * 16 + lm) * DH_;
  bf16x16 q0, q1;
  {
    bf16x8 l0 = *(const bf16x8*)(qbase + half * 16);
    bf16x8 h0 = *(const bf16x8*)(qbase + 8 + half * 16);
    bf16x8 l1 = *(const bf16x8*)(qbase + 32 + half * 16);
    bf16x8 h1 = *(const bf16x8*)(qbase + 40 + half * 16);
#pragma unroll
    for (int j = 0; j < 8; ++j) {
      q0[j] = l0[j]; q0[8 + j] = h0[j];
      q1[j] = l1[j]; q1[8 + j] = h1[j];
    }
  }

  f32x8 o0 = {}, o1 = {}, o2 = {}, o3 = {};
  float mrow[8], lrow[8];
#pragma unroll
  for (int r = 0; r < 8; ++r) { mrow[r] = -1e30f; lrow[r] = 0.f; }

  const bf16_t* kg = K + (size_t)bh * T_ * DH_;   // [t][dh]
  const bf16_t* vg = Vt + (size_t)bh * DH_ * T_;  // [dh][t]

  const int NCH = T_ / 32;  // 64 chunks
  if (wv == 0) {
    tdm_load_2d(lds_base,        kg, 64, 32, 64,  64, 32);          // K chunk 0
    tdm_load_2d(lds_base + 8192, vg, 32, 64, T_,  32, 64);          // V chunk 0
    __builtin_amdgcn_s_wait_tensorcnt(0);
  }
  __syncthreads();

  for (int i = 0; i < NCH; ++i) {
    const int cur = i & 1;
    // prefetch next chunk into the other buffer via TDM (overlaps compute)
    if (wv == 0 && i + 1 < NCH) {
      tdm_load_2d(lds_base + (cur ^ 1) * 4096, kg + (size_t)(i + 1) * 32 * DH_,
                  64, 32, 64, 64, 32);
      tdm_load_2d(lds_base + 8192 + (cur ^ 1) * 4096, vg + (size_t)(i + 1) * 32,
                  32, 64, T_, 32, 64);
    }

    // S = Q * K^T from LDS  (two 16x16 score tiles)
    const bf16_t* kc = sK0 + cur * 2048;
    const bf16_t* kf0 = kc + lm * 64 + half * 16;
    const bf16_t* kf1 = kc + (16 + lm) * 64 + half * 16;
    f32x8 s0 = {}, s1 = {};
    s0 = wmma_bf16(q0, *(const bf16x16*)kf0, s0);
    s0 = wmma_bf16(q1, *(const bf16x16*)(kf0 + 32), s0);
    s1 = wmma_bf16(q0, *(const bf16x16*)kf1, s1);
    s1 = wmma_bf16(q1, *(const bf16x16*)(kf1 + 32), s1);

    // online softmax (scale = 1/sqrt(64))
    float mx[8];
#pragma unroll
    for (int r = 0; r < 8; ++r) {
      s0[r] *= 0.125f; s1[r] *= 0.125f;
      mx[r] = fmaxf(s0[r], s1[r]);
    }
#pragma unroll
    for (int off = 8; off >= 1; off >>= 1)
#pragma unroll
      for (int r = 0; r < 8; ++r)
        mx[r] = fmaxf(mx[r], __shfl_xor(mx[r], off, 32));
#pragma unroll
    for (int r = 0; r < 8; ++r) {
      float mn = fmaxf(mrow[r], mx[r]);
      float al = __expf(mrow[r] - mn);
      mrow[r] = mn;
      float p0 = __expf(s0[r] - mn);
      float p1 = __expf(s1[r] - mn);
      s0[r] = p0; s1[r] = p1;
      lrow[r] = lrow[r] * al + p0 + p1;
      o0[r] *= al; o1[r] *= al; o2[r] *= al; o3[r] *= al;
    }

    // transpose P through LDS into A-operand layout (wave-private slot)
#pragma unroll
    for (int r = 0; r < 8; ++r) {
      sP[(half * 8 + r) * 32 + lm]      = (bf16_t)s0[r];
      sP[(half * 8 + r) * 32 + lm + 16] = (bf16_t)s1[r];
    }
    asm volatile("s_wait_dscnt 0" ::: "memory");
    bf16x8 plo = *(const bf16x8*)(sP + lm * 32 + half * 16);
    bf16x8 phi = *(const bf16x8*)(sP + lm * 32 + half * 16 + 8);
    bf16x16 ap;
#pragma unroll
    for (int j = 0; j < 8; ++j) { ap[j] = plo[j]; ap[8 + j] = phi[j]; }

    // O += P * V from LDS (4 output d-blocks)
    const bf16_t* vc = sV0 + cur * 2048;
    const bf16_t* vf = vc + lm * 32 + half * 16;
    o0 = wmma_bf16(ap, *(const bf16x16*)(vf),            o0);
    o1 = wmma_bf16(ap, *(const bf16x16*)(vf + 16 * 32),  o1);
    o2 = wmma_bf16(ap, *(const bf16x16*)(vf + 32 * 32),  o2);
    o3 = wmma_bf16(ap, *(const bf16x16*)(vf + 48 * 32),  o3);

    if (wv == 0 && i + 1 < NCH) __builtin_amdgcn_s_wait_tensorcnt(0);
    __syncthreads();  // publish next chunk / release current buffers
  }

#pragma unroll
  for (int off = 8; off >= 1; off >>= 1)
#pragma unroll
    for (int r = 0; r < 8; ++r)
      lrow[r] += __shfl_xor(lrow[r], off, 32);

  const int b = bh >> 4, h = bh & (H_ - 1);
#pragma unroll
  for (int r = 0; r < 8; ++r) {
    float inv = 1.f / lrow[r];
    int t = qt * 16 + half * 8 + r;
    size_t base = ((size_t)b * T_ + t) * D_ + h * DH_ + lm;
    O[base]      = (bf16_t)(o0[r] * inv);
    O[base + 16] = (bf16_t)(o1[r] * inv);
    O[base + 32] = (bf16_t)(o2[r] * inv);
    O[base + 48] = (bf16_t)(o3[r] * inv);
  }
}

// ---------------- output projection GEMM (wave tile 16x64) ----------------
__global__ __launch_bounds__(256) void out_gemm_kernel(
    const bf16_t* __restrict__ Ob, const bf16_t* __restrict__ wT,
    const float* __restrict__ bias, float* __restrict__ out) {
  const int lane = threadIdx.x & 31, wv = threadIdx.x >> 5;
  const int half = lane >> 4, lm = lane & 15;
  const int n0 = (blockIdx.x * 8 + wv) * 64;   // 0..1023, step 64
  const int m0 = blockIdx.y * 16;
  const bf16_t* arow = Ob + (size_t)(m0 + lm) * D_;
  const bf16_t* brow = wT + (size_t)(n0 + lm) * D_;

  f32x8 acc[4] = {{}, {}, {}, {}};
  for (int kk = 0; kk < D_; kk += 32) {
    bf16x8 alo = *(const bf16x8*)(arow + kk + half * 16);
    bf16x8 ahi = *(const bf16x8*)(arow + kk + half * 16 + 8);
    bf16x16 a;
#pragma unroll
    for (int j = 0; j < 8; ++j) { a[j] = alo[j]; a[8 + j] = ahi[j]; }
    __builtin_prefetch(arow + kk + 128, 0, 3);
#pragma unroll
    for (int cb = 0; cb < 4; ++cb) {
      bf16x16 b = *(const bf16x16*)(brow + (size_t)cb * 16 * D_ + kk + half * 16);
      acc[cb] = wmma_bf16(a, b, acc[cb]);
    }
  }

#pragma unroll
  for (int cb = 0; cb < 4; ++cb) {
    const int col = n0 + cb * 16 + lm;
    const float bv = bias[col];
#pragma unroll
    for (int r = 0; r < 8; ++r) {
      int mrow = m0 + half * 8 + r;
      out[(size_t)mrow * D_ + col] = acc[cb][r] + bv;
    }
  }
}

// ---------------- launch ----------------

extern "C" void kernel_launch(void* const* d_in, const int* in_sizes, int n_in,
                              void* d_out, int out_size, void* d_ws, size_t ws_size,
                              hipStream_t stream) {
  const float* x    = (const float*)d_in[0];
  const float* Wqkv = (const float*)d_in[1];
  const float* bqkv = (const float*)d_in[2];
  const float* Wout = (const float*)d_in[3];
  const float* bout = (const float*)d_in[4];
  float* out = (float*)d_out;

  char* ws = (char*)d_ws;
  size_t off = 0;
  auto carve = [&](size_t bytes) -> char* {
    char* p = ws + off;
    off += (bytes + 255) & ~(size_t)255;
    return p;
  };
  const size_t NTOK = (size_t)B_ * T_;  // 8192
  bf16_t* xb    = (bf16_t*)carve(NTOK * D_ * 2);
  bf16_t* wqkvT = (bf16_t*)carve((size_t)3 * D_ * D_ * 2);
  bf16_t* woutT = (bf16_t*)carve((size_t)D_ * D_ * 2);
  bf16_t* Qb    = (bf16_t*)carve(NTOK * D_ * 2);
  bf16_t* Kb    = (bf16_t*)carve(NTOK * D_ * 2);
  bf16_t* Vt    = (bf16_t*)carve(NTOK * D_ * 2);
  bf16_t* Ob    = (bf16_t*)carve(NTOK * D_ * 2);

  int nx = (int)(NTOK * D_);
  cvt_kernel<<<(nx + 255) / 256, 256, 0, stream>>>(x, xb, nx);
  cvtT_kernel<<<(3 * D_ * D_ + 255) / 256, 256, 0, stream>>>(Wqkv, wqkvT, 3 * D_);
  cvtT_kernel<<<(D_ * D_ + 255) / 256, 256, 0, stream>>>(Wout, woutT, D_);

  // 48 N-supertiles (64 wide) / 8 waves, 512 M-tiles
  qkv_gemm_kernel<<<dim3(6, 512), 256, 0, stream>>>(xb, wqkvT, bqkv, Qb, Kb, Vt);

  // B*H*T/16 = 8192 wave-tiles, 4 waves per block (shared b,h per block)
  attn_kernel<<<2048, 128, 0, stream>>>(Qb, Kb, Vt, Ob);

  // 16 N-supertiles / 8 waves, 512 M-tiles
  out_gemm_kernel<<<dim3(2, 512), 256, 0, stream>>>(Ob, woutT, bout, out);
}
